// MotionEstimator_47141561041516
// MI455X (gfx1250) — matrix operand order repaired and checked by reading, not compiled, not run on hardware
//
#include <hip/hip_runtime.h>
#include <hip/hip_bf16.h>

typedef __attribute__((ext_vector_type(16))) __bf16 v16bf;
typedef __attribute__((ext_vector_type(8)))  __bf16 v8bf;
typedef __attribute__((ext_vector_type(8)))  float  v8f;

#define HH  128
#define WWI 224
#define HWN (HH * WWI)   // 28672
#define BB  4
#define C0P 480          // conv1 input channels padded (469 -> 480)
// stored channel order of the concat tensor (for alignment):
//   f0 @ 0..127, f1 @ 128..255, volume @ 256..336, last_feat @ 337..464,
//   last_flow @ 465..468, zero-pad @ 469..479
// (conv1's packed weights are permuted to match, so the math is unchanged)

// ---------------------------------------------------------------- utilities
__global__ void zero_f32_kernel(float* __restrict__ p, long n) {
    long i = (long)blockIdx.x * blockDim.x + threadIdx.x;
    if (i < n) p[i] = 0.0f;
}

// zero the padding channels 469..479 of the NHWC concat tensor
__global__ void pad_x0_kernel(__bf16* __restrict__ x0) {
    long idx = (long)blockIdx.x * blockDim.x + threadIdx.x;
    long total = (long)BB * HWN * 11;
    if (idx >= total) return;
    int c = (int)(idx % 11);
    long pp = idx / 11;                       // b*HWN + p
    x0[pp * C0P + 469 + c] = (__bf16)0.0f;
}

// ------------------------------------------------------------- softsplat add
// acc layout: [B][129][HWN] f32 (channel 128 = "ones" normalizer)
__global__ void splat_add_kernel(const float* __restrict__ feat,
                                 const float* __restrict__ flow, int flow_off,
                                 float* __restrict__ acc) {
    long idx = (long)blockIdx.x * blockDim.x + threadIdx.x;
    long total = (long)BB * 129 * HWN;
    if (idx >= total) return;
    int p = (int)(idx % HWN);
    int c = (int)((idx / HWN) % 129);
    int b = (int)(idx / (129L * HWN));
    int x = p % WWI, y = p / WWI;
    float fx = flow[(((long)b * 4 + flow_off)     * HWN) + p] * 0.125f + (float)x;
    float fy = flow[(((long)b * 4 + flow_off + 1) * HWN) + p] * 0.125f + (float)y;
    float v = (c < 128) ? feat[(((long)b * 128 + c) * HWN) + p] : 1.0f;
    float x0 = floorf(fx), y0 = floorf(fy);
    float* accb = acc + ((long)b * 129 + c) * HWN;
#pragma unroll
    for (int dx = 0; dx < 2; ++dx) {
#pragma unroll
        for (int dy = 0; dy < 2; ++dy) {
            float xi = x0 + (float)dx, yi = y0 + (float)dy;
            float wgt = (1.0f - fabsf(fx - xi)) * (1.0f - fabsf(fy - yi));
            int xii = (int)xi, yii = (int)yi;
            if (xii >= 0 && xii < WWI && yii >= 0 && yii < HH)
                atomicAdd(accb + yii * WWI + xii, v * wgt);
        }
    }
}

// -------------------------- normalize + write bf16 into NHWC concat tensor
__global__ void normalize_kernel(const float* __restrict__ acc,
                                 __bf16* __restrict__ xdst, int ch_off) {
    long idx = (long)blockIdx.x * blockDim.x + threadIdx.x;
    long total = (long)BB * 128 * HWN;
    if (idx >= total) return;
    int p = (int)(idx % HWN);
    int c = (int)((idx / HWN) % 128);
    int b = (int)(idx / (128L * HWN));
    float norm = acc[((long)b * 129 + 128) * HWN + p];
    if (norm == 0.0f) norm = 1.0f;
    float v = acc[((long)b * 129 + c) * HWN + p] / norm;
    xdst[((long)b * HWN + p) * C0P + ch_off + c] = (__bf16)v;
}

// ----------------------------------------------- correlation + leaky (bf16)
// NHWC: reads f0 at ch 0.., f1 at ch 128.. (32B-aligned), writes ch 256..336
__global__ void corr_kernel(const __bf16* __restrict__ xsrc,
                            __bf16* __restrict__ xdst) {
    long idx = (long)blockIdx.x * blockDim.x + threadIdx.x;
    long total = (long)BB * 81 * HWN;
    if (idx >= total) return;
    int p = (int)(idx % HWN);
    int d = (int)((idx / HWN) % 81);
    int b = (int)(idx / (81L * HWN));
    int x = p % WWI, y = p / WWI;
    int dy = d / 9 - 4, dx = d % 9 - 4;
    int y2 = y + dy, x2 = x + dx;
    float s = 0.0f;
    if (x2 >= 0 && x2 < WWI && y2 >= 0 && y2 < HH) {
        const v16bf* F0 = (const v16bf*)(xsrc + ((long)b * HWN + p) * C0P);
        const v16bf* F1 = (const v16bf*)(xsrc + ((long)b * HWN + y2 * WWI + x2) * C0P + 128);
#pragma unroll
        for (int cc = 0; cc < 8; ++cc) {
            v16bf u = F0[cc], w = F1[cc];
#pragma unroll
            for (int e = 0; e < 16; ++e)
                s += (float)u[e] * (float)w[e];
        }
        s *= (1.0f / 128.0f);
    }
    s = (s >= 0.0f) ? s : 0.1f * s;
    xdst[((long)b * HWN + p) * C0P + 256 + d] = (__bf16)s;
}

// ----------------------------- copy last_feat (ch 337..464) & last_flow tail
__global__ void copy_tail_kernel(const float* __restrict__ last_feat,
                                 const float* __restrict__ last_flow,
                                 __bf16* __restrict__ xdst) {
    long idx = (long)blockIdx.x * blockDim.x + threadIdx.x;
    long total = (long)BB * 132 * HWN;
    if (idx >= total) return;
    int p = (int)(idx % HWN);
    int c = (int)((idx / HWN) % 132);
    int b = (int)(idx / (132L * HWN));
    float v; int ch;
    if (c < 128) { v = last_feat[(((long)b * 128 + c) * HWN) + p]; ch = 337 + c; }
    else         { v = last_flow[(((long)b * 4 + (c - 128)) * HWN) + p]; ch = 465 + (c - 128); }
    xdst[((long)b * HWN + p) * C0P + ch] = (__bf16)v;
}

// -------------------------------------------- weight packing into A-fragments
// k ordering is t-major: k = t*CINP + ci_stored, t = r*KK + s.
// REMAP (conv1 only): stored channel -> reference channel:
//   [0,128)   -> 81+c   (f0)
//   [128,256) -> 209-128+c = 81+c? no: c-128+209  (f1)
//   [256,337) -> c-256  (volume)
//   [337,480) -> c      (last_feat/last_flow; pad >=469 zero)
// packed layout: [mtile16][kstep][lane(32)][elem(16)] bf16
// A 16x32 bf16 fragment as packed here (B loads use same k order): row M = lane&15 ; K = 16*(lane>>4) + e
template <int CINP, int CIN, int COUT, int KK, int MT16, bool REMAP>
__global__ void pack_w_kernel(const float* __restrict__ w, __bf16* __restrict__ wpk) {
    constexpr int KK2    = KK * KK;
    constexpr int KSTEPS = (CINP * KK2) / 32;
    long total = (long)MT16 * KSTEPS * 32 * 16;
    long idx = (long)blockIdx.x * blockDim.x + threadIdx.x;
    if (idx >= total) return;
    int e     = (int)(idx & 15);
    int lane  = (int)((idx >> 4) & 31);
    long rest = idx >> 9;
    int kstep = (int)(rest % KSTEPS);
    int mtile = (int)(rest / KSTEPS);
    int m  = mtile * 16 + (lane & 15);
    int kl = ((lane >> 4) * 16) + e;
    int k  = kstep * 32 + kl;
    int t  = k / CINP;
    int ci = k % CINP;               // stored-channel index
    if (REMAP) {                      // stored -> reference channel
        if      (ci < 128) ci = 81 + ci;
        else if (ci < 256) ci = (ci - 128) + 209;
        else if (ci < 337) ci = ci - 256;
        /* else: ci unchanged (last_feat/last_flow/pad) */
    }
    float v = 0.0f;
    if (m < COUT && ci < CIN)
        v = w[(((long)m * CIN + ci) * KK2) + t];
    wpk[idx] = (__bf16)v;
}

// ------------------------------------------------- implicit-GEMM conv (WMMA)
// NHWC bf16 activations. Wave tile: 32 couts x 32 pixels (2 A x 2 B frags,
// 4 WMMAs / K-step). Block = 8 waves arranged MW (couts) x NW (pixels).
// grid: (HWN/(NW*32), ceil(COUT/(MW*32)), B)
template <int CINP, int COUT, int KK, int MW, bool LEAKY, bool OUTBF, bool OUTF>
__global__ __launch_bounds__(256)
void conv_wmma_kernel(const __bf16* __restrict__ xin,   // [B][HWN][CINP]
                      const __bf16* __restrict__ wpk,
                      const float* __restrict__ bias,
                      __bf16* __restrict__ ybf,          // [B][HWN][COUT]
                      float* __restrict__ yf) {          // [B][COUT][HWN]
    constexpr int NW     = 8 / MW;
    constexpr int KSTEPS = (CINP * KK * KK) / 32;
    constexpr int CS     = CINP / 32;     // K-steps per filter tap
    constexpr int PAD    = KK / 2;

    const int lane  = threadIdx.x & 31;
    const int wv    = threadIdx.x >> 5;
    const int mwv   = wv % MW;
    const int nwv   = wv / MW;
    const int b     = blockIdx.z;
    const int khalf = lane >> 4;          // 0 or 1
    const long bHW  = (long)b * HWN;

    // two pixel tiles per wave
    int pcoord[2], xc[2], yc[2];
#pragma unroll
    for (int nt = 0; nt < 2; ++nt) {
        pcoord[nt] = blockIdx.x * (NW * 32) + nwv * 32 + nt * 16 + (lane & 15);
        xc[nt] = pcoord[nt] % WWI;
        yc[nt] = pcoord[nt] / WWI;
    }

    v8f acc00 = {}, acc01 = {}, acc10 = {}, acc11 = {};
    const int mt_g0 = blockIdx.y * (MW * 2) + mwv * 2;
    const __bf16* wp0 = wpk + ((long)(mt_g0 + 0) * KSTEPS) * 512 + lane * 16;
    const __bf16* wp1 = wpk + ((long)(mt_g0 + 1) * KSTEPS) * 512 + lane * 16;

    for (int ks = 0; ks < KSTEPS; ++ks) {
        const int t  = ks / CS;                       // filter tap (fixed per K-step)
        const int c0 = (ks - t * CS) * 32 + khalf * 16;
        const int dy = (KK == 1) ? 0 : (t / KK - PAD);
        const int dx = (KK == 1) ? 0 : (t % KK - PAD);

        // A fragments: one 32B vector load each (broadcast across waves via WGP$)
        v16bf a0 = *(const v16bf*)(wp0 + (long)ks * 512);
        v16bf a1 = *(const v16bf*)(wp1 + (long)ks * 512);
        __builtin_prefetch((const void*)(wp0 + (long)(ks + 1) * 512), 0, 1);

        // B fragments: 16 consecutive channels at a shifted pixel = one 32B load
        v16bf bf0 = {}, bf1 = {};
        {
            int px = xc[0] + dx, py = yc[0] + dy;
            if (px >= 0 && px < WWI && py >= 0 && py < HH)
                bf0 = *(const v16bf*)(xin + (bHW + py * WWI + px) * CINP + c0);
        }
        {
            int px = xc[1] + dx, py = yc[1] + dy;
            if (px >= 0 && px < WWI && py >= 0 && py < HH)
                bf1 = *(const v16bf*)(xin + (bHW + py * WWI + px) * CINP + c0);
        }

        acc00 = __builtin_amdgcn_wmma_f32_16x16x32_bf16(false, a0, false, bf0, (short)0, acc00, false, false);
        acc01 = __builtin_amdgcn_wmma_f32_16x16x32_bf16(false, a0, false, bf1, (short)0, acc01, false, false);
        acc10 = __builtin_amdgcn_wmma_f32_16x16x32_bf16(false, a1, false, bf0, (short)0, acc10, false, false);
        acc11 = __builtin_amdgcn_wmma_f32_16x16x32_bf16(false, a1, false, bf1, (short)0, acc11, false, false);
    }

    // epilogue: C/D layout: VGPR v -> M = 8*(lane>>4) + v within a 16-row tile
#pragma unroll
    for (int mt = 0; mt < 2; ++mt) {
#pragma unroll
        for (int nt = 0; nt < 2; ++nt) {
            const v8f a = (mt == 0) ? (nt == 0 ? acc00 : acc01)
                                    : (nt == 0 ? acc10 : acc11);
            const int m0 = (mt_g0 + mt) * 16 + khalf * 8;
            const int p  = pcoord[nt];
            if constexpr (OUTBF) {   // NHWC: 8 consecutive channels -> one 16B store
                if (m0 < COUT) {     // COUT % 32 == 0 here, so m0+8 <= COUT
                    v8bf ov;
#pragma unroll
                    for (int v = 0; v < 8; ++v) {
                        float val = a[v] + bias[m0 + v];
                        if (LEAKY) val = (val >= 0.0f) ? val : 0.1f * val;
                        ov[v] = (__bf16)val;
                    }
                    *(v8bf*)(ybf + (bHW + p) * COUT + m0) = ov;
                }
            }
            if constexpr (OUTF) {    // NCHW f32 (final outputs)
#pragma unroll
                for (int v = 0; v < 8; ++v) {
                    int m = m0 + v;
                    if (m < COUT) {
                        float val = a[v] + bias[m];
                        if (LEAKY) val = (val >= 0.0f) ? val : 0.1f * val;
                        yf[((long)b * COUT + m) * HWN + p] = val;
                    }
                }
            }
        }
    }
}

// ---------------------------------------------------------------------------
static inline long cdivl(long a, long b) { return (a + b - 1) / b; }

extern "C" void kernel_launch(void* const* d_in, const int* in_sizes, int n_in,
                              void* d_out, int out_size, void* d_ws, size_t ws_size,
                              hipStream_t stream) {
    const float* feat0     = (const float*)d_in[0];
    const float* feat1     = (const float*)d_in[1];
    const float* last_feat = (const float*)d_in[2];
    const float* last_flow = (const float*)d_in[3];
    const float* w1 = (const float*)d_in[4];  const float* b1 = (const float*)d_in[5];
    const float* w2 = (const float*)d_in[6];  const float* b2 = (const float*)d_in[7];
    const float* w3 = (const float*)d_in[8];  const float* b3 = (const float*)d_in[9];
    const float* w4 = (const float*)d_in[10]; const float* b4 = (const float*)d_in[11];
    const float* w5 = (const float*)d_in[12]; const float* b5 = (const float*)d_in[13];
    const float* w6 = (const float*)d_in[14]; const float* b6 = (const float*)d_in[15];

    char* ws = (char*)d_ws;
    size_t off = 0;
    auto take = [&](size_t bytes) { char* p = ws + off; off = (off + bytes + 255) & ~(size_t)255; return p; };

    float*  acc0 = (float*)take((size_t)BB * 129 * HWN * 4);
    float*  acc1 = (float*)take((size_t)BB * 129 * HWN * 4);
    __bf16* bufA = (__bf16*)take((size_t)BB * HWN * C0P * 2);  // NHWC, padded
    __bf16* bufB = (__bf16*)take((size_t)BB * HWN * 320 * 2);  // NHWC
    __bf16* wp1 = (__bf16*)take((size_t)20 * 15 * 512 * 2);    // 480(469)->320, 1x1
    __bf16* wp2 = (__bf16*)take((size_t)16 * 90 * 512 * 2);    // 320->256, 3x3
    __bf16* wp3 = (__bf16*)take((size_t)16 * 72 * 512 * 2);    // 256->224 (padded M)
    __bf16* wp4 = (__bf16*)take((size_t)12 * 63 * 512 * 2);    // 224->192
    __bf16* wp5 = (__bf16*)take((size_t)8  * 54 * 512 * 2);    // 192->128
    __bf16* wp6 = (__bf16*)take((size_t)2  * 36 * 512 * 2);    // 128->4

    float* flow_out = (float*)d_out;                           // B*4*HWN (NCHW)
    float* feat_out = (float*)d_out + (size_t)BB * 4 * HWN;    // B*128*HWN (NCHW)

    const int T = 256;

    // 1) zero splat accumulators (acc0 & acc1 contiguous)
    {
        long n = 2L * BB * 129 * HWN;
        zero_f32_kernel<<<(unsigned)cdivl(n, T), T, 0, stream>>>(acc0, n);
    }
    // 2) forward splat (atomic scatter-add)
    {
        long n = (long)BB * 129 * HWN;
        splat_add_kernel<<<(unsigned)cdivl(n, T), T, 0, stream>>>(feat0, last_flow, 0, acc0);
        splat_add_kernel<<<(unsigned)cdivl(n, T), T, 0, stream>>>(feat1, last_flow, 2, acc1);
    }
    // 3) normalize -> NHWC bf16: f0 @ ch 0, f1 @ ch 128 (aligned)
    {
        long n = (long)BB * 128 * HWN;
        normalize_kernel<<<(unsigned)cdivl(n, T), T, 0, stream>>>(acc0, bufA, 0);
        normalize_kernel<<<(unsigned)cdivl(n, T), T, 0, stream>>>(acc1, bufA, 128);
    }
    // 4) correlation volume + leaky -> channels 256..336
    {
        long n = (long)BB * 81 * HWN;
        corr_kernel<<<(unsigned)cdivl(n, T), T, 0, stream>>>(bufA, bufA);
    }
    // 5) tails: last_feat/last_flow -> ch 337..468 ; zero pad ch 469..479
    {
        long n = (long)BB * 132 * HWN;
        copy_tail_kernel<<<(unsigned)cdivl(n, T), T, 0, stream>>>(last_feat, last_flow, bufA);
        long nz = (long)BB * HWN * 11;
        pad_x0_kernel<<<(unsigned)cdivl(nz, T), T, 0, stream>>>(bufA);
    }
    // 6) pack weights into WMMA A-fragment layout (t-major K; conv1 remapped)
    pack_w_kernel<C0P, 469, 320, 1, 20, true ><<<(unsigned)cdivl(20L * 15 * 512, T), T, 0, stream>>>(w1, wp1);
    pack_w_kernel<320, 320, 256, 3, 16, false><<<(unsigned)cdivl(16L * 90 * 512, T), T, 0, stream>>>(w2, wp2);
    pack_w_kernel<256, 256, 224, 3, 16, false><<<(unsigned)cdivl(16L * 72 * 512, T), T, 0, stream>>>(w3, wp3);
    pack_w_kernel<224, 224, 192, 3, 12, false><<<(unsigned)cdivl(12L * 63 * 512, T), T, 0, stream>>>(w4, wp4);
    pack_w_kernel<192, 192, 128, 3, 8,  false><<<(unsigned)cdivl(8L  * 54 * 512, T), T, 0, stream>>>(w5, wp5);
    pack_w_kernel<128, 128, 4,   3, 2,  false><<<(unsigned)cdivl(2L  * 36 * 512, T), T, 0, stream>>>(w6, wp6);

    // 7) conv stack (implicit GEMM, bf16 WMMA, f32 accumulate)
    // MW chosen per layer to minimize L2 re-reads (redundancy = taps * gridY)
    const dim3 blk(256);
    conv_wmma_kernel<C0P, 320, 1, 2, true,  true,  false><<<dim3(HWN/128, 5, BB), blk, 0, stream>>>(bufA, wp1, b1, bufB, nullptr);
    conv_wmma_kernel<320, 256, 3, 8, true,  true,  false><<<dim3(HWN/32,  1, BB), blk, 0, stream>>>(bufB, wp2, b2, bufA, nullptr);
    conv_wmma_kernel<256, 224, 3, 8, true,  true,  false><<<dim3(HWN/32,  1, BB), blk, 0, stream>>>(bufA, wp3, b3, bufB, nullptr);
    conv_wmma_kernel<224, 192, 3, 2, true,  true,  false><<<dim3(HWN/128, 3, BB), blk, 0, stream>>>(bufB, wp4, b4, bufA, nullptr);
    conv_wmma_kernel<192, 128, 3, 4, true,  true,  true ><<<dim3(HWN/64,  1, BB), blk, 0, stream>>>(bufA, wp5, b5, bufB, feat_out);
    conv_wmma_kernel<128, 4,   3, 1, false, false, true ><<<dim3(HWN/256, 1, BB), blk, 0, stream>>>(bufB, wp6, b6, nullptr, flow_out);
}
